// ISTRPredictor_35424890258079
// MI455X (gfx1250) — compile-verified, matrix operand order checked by reading, not run on hardware
//
#include <hip/hip_runtime.h>
#include <math.h>

// ---------------- model constants ----------------
#define Bq 8
#define Sq_ 1024
#define Dm 512
#define Hh 8
#define HDc 64
#define DFFc 2048

// ---------------- WMMA types ----------------
typedef __bf16 bf16_t;
typedef bf16_t v16bf __attribute__((ext_vector_type(16)));
typedef float v8f __attribute__((ext_vector_type(8)));

union Frag { uint4 u[2]; v16bf v; };

__device__ __forceinline__ unsigned short f2bf(float f) {
  union { float f; unsigned u; } v; v.f = f;
  unsigned r = v.u + 0x7FFFu + ((v.u >> 16) & 1u);
  return (unsigned short)(r >> 16);
}

__device__ __forceinline__ float gelu_f(float x) {
  float x3 = x * x * x;
  return 0.5f * x * (1.0f + tanhf(0.7978845608028654f * (x + 0.044715f * x3)));
}

// async copy 16B global -> LDS (ASYNCcnt-tracked, no VGPR round-trip)
__device__ __forceinline__ void async_cp16(const unsigned short* g, unsigned short* l) {
  unsigned lds = (unsigned)(uintptr_t)l;   // low 32 bits = LDS byte address
  asm volatile("global_load_async_to_lds_b128 %0, %1, off"
               :: "v"(lds), "v"(g) : "memory");
}
__device__ __forceinline__ void wait_async0() {
  asm volatile("s_wait_asynccnt 0x0" ::: "memory");
}
__device__ __forceinline__ void wait_async4() {
  asm volatile("s_wait_asynccnt 0x4" ::: "memory");
}

// =====================================================================
// bf16 WMMA GEMM:  C[M,N] = A[M,K] * W[K,N] + bias, optional GELU.
// A: bf16 row-major [M,K]; Wt: bf16 [N,K] (pre-transposed weights).
// Block 256 thr = 8 waves; tile 128x128, BK=32; wave tile 32x64.
// Double-buffered LDS staged with GLOBAL_LOAD_ASYNC_TO_LDS_B128;
// last tile peeled so the steady-state loop body is branch-free.
// =====================================================================
__global__ __launch_bounds__(256) void gemm_bf16_kernel(
    const unsigned short* __restrict__ A,
    const unsigned short* __restrict__ Wt,
    const float* __restrict__ bias,
    float* __restrict__ outF,
    unsigned short* __restrict__ outB,
    int M, int N, int K, int act)
{
  constexpr int ST = 48;  // padded LDS stride (bf16 elems): 96B, 16B-aligned rows
  __shared__ unsigned short As[2][128 * ST];
  __shared__ unsigned short Bs[2][128 * ST];
  const int tid   = threadIdx.x;
  const int wave  = tid >> 5;
  const int lane  = tid & 31;
  const int l15   = lane & 15;
  const int lhalf = lane >> 4;
  const int koff  = lhalf * 8;
  const int bm = blockIdx.x * 128;
  const int bn = blockIdx.y * 128;
  const int wm = (wave >> 1) * 32;
  const int wn = (wave & 1) * 64;

  v8f zf = {};
  v8f c[2][4];
#pragma unroll
  for (int i = 0; i < 2; ++i)
#pragma unroll
    for (int j = 0; j < 4; ++j) c[i][j] = zf;

  // per-thread staging coordinates: 2 chunks of 8 bf16 per matrix,
  // running global pointers advanced by BK each issue.
  const int r0 = tid >> 2,           c0 = (tid & 3) * 8;          // chunk 0
  const int r1 = (tid + 256) >> 2,   c1 = ((tid + 256) & 3) * 8;  // chunk 1
  const unsigned short* gA0 = A  + (size_t)(bm + r0) * K + c0;
  const unsigned short* gA1 = A  + (size_t)(bm + r1) * K + c1;
  const unsigned short* gB0 = Wt + (size_t)(bn + r0) * K + c0;
  const unsigned short* gB1 = Wt + (size_t)(bn + r1) * K + c1;
  unsigned short* lA0[2] = { &As[0][r0 * ST + c0], &As[1][r0 * ST + c0] };
  unsigned short* lA1[2] = { &As[0][r1 * ST + c1], &As[1][r1 * ST + c1] };
  unsigned short* lB0[2] = { &Bs[0][r0 * ST + c0], &Bs[1][r0 * ST + c0] };
  unsigned short* lB1[2] = { &Bs[0][r1 * ST + c1], &Bs[1][r1 * ST + c1] };

  auto issue = [&](int buf) {
    async_cp16(gA0, lA0[buf]);  gA0 += 32;
    async_cp16(gA1, lA1[buf]);  gA1 += 32;
    async_cp16(gB0, lB0[buf]);  gB0 += 32;
    async_cp16(gB1, lB1[buf]);  gB1 += 32;
  };
  auto compute = [&](int cur) {
    Frag a[2], b[4];
#pragma unroll
    for (int i = 0; i < 2; ++i) {
      const unsigned short* p = &As[cur][(wm + i * 16 + l15) * ST];
      a[i].u[0] = *(const uint4*)(p + koff);
      a[i].u[1] = *(const uint4*)(p + koff + 16);
    }
#pragma unroll
    for (int j = 0; j < 4; ++j) {
      const unsigned short* p = &Bs[cur][(wn + j * 16 + l15) * ST];
      b[j].u[0] = *(const uint4*)(p + koff);
      b[j].u[1] = *(const uint4*)(p + koff + 16);
    }
#pragma unroll
    for (int i = 0; i < 2; ++i)
#pragma unroll
      for (int j = 0; j < 4; ++j)
        c[i][j] = __builtin_amdgcn_wmma_f32_16x16x32_bf16(
            false, a[i].v, false, b[j].v, (short)0, c[i][j], false, false);
  };

  const int nK = K >> 5;
  issue(0);
  for (int kt = 0; kt < nK - 1; ++kt) {   // branch-free steady state
    const int cur = kt & 1;
    issue(cur ^ 1);       // prefetch next tile (async, overlaps compute)
    wait_async4();        // own copies of current tile complete
    __syncthreads();
    compute(cur);
    __syncthreads();
  }
  wait_async0();          // peeled last tile
  __syncthreads();
  compute((nK - 1) & 1);

#pragma unroll
  for (int i = 0; i < 2; ++i) {
#pragma unroll
    for (int j = 0; j < 4; ++j) {
      int col = bn + wn + j * 16 + l15;
      float bb = bias ? bias[col] : 0.0f;
#pragma unroll
      for (int r = 0; r < 8; ++r) {
        int row = bm + wm + i * 16 + r + 8 * lhalf;
        float v = c[i][j][r] + bb;
        if (act == 1) v = gelu_f(v);
        size_t idx = (size_t)row * N + col;
        if (outF) outF[idx] = v;
        if (outB) outB[idx] = f2bf(v);
      }
    }
  }
}

// =====================================================================
// Banded flash attention. Q,K,V,O: bf16 [B*S, D], head h in cols h*64..
// Grid: (Sq/64, B*H); 128 threads = 4 waves; wave owns 16 q-rows.
// Per 32-key tile: 4 WMMA for Q*K^T, online softmax, 4 WMMA for P*V.
// Q/K tiles staged with async-to-LDS; V transposed manually.
// =====================================================================
__global__ __launch_bounds__(128) void banded_attn_kernel(
    const unsigned short* __restrict__ Q,
    const unsigned short* __restrict__ Kc,
    const unsigned short* __restrict__ Vc,
    unsigned short* __restrict__ O,
    int Sq, int Sk, int bw)
{
  constexpr int STK = 72;  // stride for 64-wide hd rows (144B, aligned)
  constexpr int STV = 48;  // stride for 32-wide key rows (96B, aligned)
  __shared__ unsigned short Qs[64 * STK];
  __shared__ unsigned short Ks[32 * STK];
  __shared__ unsigned short Vs[64 * STV];      // transposed: [hd][key]
  __shared__ unsigned short Ps[4][16 * STV];   // per-wave P tile [q][key]

  const int tid   = threadIdx.x;
  const int wave  = tid >> 5;
  const int lane  = tid & 31;
  const int l15   = lane & 15;
  const int lhalf = lane >> 4;
  const int koff  = lhalf * 8;
  const int q0 = blockIdx.x * 64;
  const int bh = blockIdx.y;
  const int bb = bh / Hh;
  const int hh = bh % Hh;

  const unsigned short* Qg = Q  + ((size_t)bb * Sq) * Dm + hh * HDc;
  const unsigned short* Kg = Kc + ((size_t)bb * Sk) * Dm + hh * HDc;
  const unsigned short* Vg = Vc + ((size_t)bb * Sk) * Dm + hh * HDc;

  // stage Q tile (64 x 64) via async copies
#pragma unroll
  for (int i = 0; i < 4; ++i) {
    int chunk = tid + i * 128;
    int r  = chunk >> 3;
    int cc = (chunk & 7) * 8;
    async_cp16(Qg + (size_t)(q0 + r) * Dm + cc, Qs + r * STK + cc);
  }
  wait_async0();
  __syncthreads();

  Frag qa[2];
  {
    const unsigned short* p = Qs + (wave * 16 + l15) * STK;
    qa[0].u[0] = *(const uint4*)(p + koff);
    qa[0].u[1] = *(const uint4*)(p + koff + 16);
    qa[1].u[0] = *(const uint4*)(p + 32 + koff);
    qa[1].u[1] = *(const uint4*)(p + 32 + koff + 16);
  }

  v8f zf = {};
  float rmax[8], rsum[8];
  v8f o[4];
#pragma unroll
  for (int r = 0; r < 8; ++r) { rmax[r] = -__builtin_inff(); rsum[r] = 0.0f; }
#pragma unroll
  for (int j = 0; j < 4; ++j) o[j] = zf;

  int ks = q0 - bw;        if (ks < 0)  ks = 0;
  int ke = q0 + 64 + bw;   if (ke > Sk) ke = Sk;

  for (int kb = ks; kb < ke; kb += 32) {
    __syncthreads();
    // K tile natural [key][hd] via async; V tile transposed into Vs[hd][key]
#pragma unroll
    for (int i = 0; i < 2; ++i) {
      int chunk = tid + i * 128;
      int r  = chunk >> 3;
      int cc = (chunk & 7) * 8;
      async_cp16(Kg + (size_t)(kb + r) * Dm + cc, Ks + r * STK + cc);
      uint4 d = *(const uint4*)(Vg + (size_t)(kb + r) * Dm + cc);
      const unsigned short* ds = (const unsigned short*)&d;
#pragma unroll
      for (int t = 0; t < 8; ++t) Vs[(cc + t) * STV + r] = ds[t];
    }
    wait_async0();
    __syncthreads();

    // scores for two 16-key sub-tiles
    Frag b0a, b0b, b1a, b1b;
    {
      const unsigned short* p0 = Ks + l15 * STK;
      const unsigned short* p1 = Ks + (16 + l15) * STK;
      b0a.u[0] = *(const uint4*)(p0 + koff);      b0a.u[1] = *(const uint4*)(p0 + koff + 16);
      b0b.u[0] = *(const uint4*)(p0 + 32 + koff); b0b.u[1] = *(const uint4*)(p0 + 32 + koff + 16);
      b1a.u[0] = *(const uint4*)(p1 + koff);      b1a.u[1] = *(const uint4*)(p1 + koff + 16);
      b1b.u[0] = *(const uint4*)(p1 + 32 + koff); b1b.u[1] = *(const uint4*)(p1 + 32 + koff + 16);
    }
    v8f s0 = __builtin_amdgcn_wmma_f32_16x16x32_bf16(false, qa[0].v, false, b0a.v, (short)0, zf, false, false);
    s0     = __builtin_amdgcn_wmma_f32_16x16x32_bf16(false, qa[1].v, false, b0b.v, (short)0, s0, false, false);
    v8f s1 = __builtin_amdgcn_wmma_f32_16x16x32_bf16(false, qa[0].v, false, b1a.v, (short)0, zf, false, false);
    s1     = __builtin_amdgcn_wmma_f32_16x16x32_bf16(false, qa[1].v, false, b1b.v, (short)0, s1, false, false);

    // online softmax (rows live across 16-lane halves)
    unsigned short* Pw = Ps[wave];
#pragma unroll
    for (int r = 0; r < 8; ++r) {
      int m = r + 8 * lhalf;
      int qrow = q0 + wave * 16 + m;
      float v0 = s0[r] * 0.125f;     // 1/sqrt(64)
      float v1 = s1[r] * 0.125f;
      int d0 = qrow - (kb + l15);       if (d0 < 0) d0 = -d0;
      int d1 = qrow - (kb + 16 + l15);  if (d1 < 0) d1 = -d1;
      if (d0 > bw) v0 = -__builtin_inff();
      if (d1 > bw) v1 = -__builtin_inff();
      float mt = fmaxf(v0, v1);
#pragma unroll
      for (int ofs = 1; ofs < 16; ofs <<= 1)
        mt = fmaxf(mt, __shfl_xor(mt, ofs, 32));
      float nm = fmaxf(rmax[r], mt);
      float sc, p0, p1;
      if (nm < -1e37f) { sc = 1.0f; p0 = 0.0f; p1 = 0.0f; }
      else {
        sc = __expf(rmax[r] - nm);
        p0 = __expf(v0 - nm);
        p1 = __expf(v1 - nm);
      }
      rmax[r] = nm;
      float ps = p0 + p1;
#pragma unroll
      for (int ofs = 1; ofs < 16; ofs <<= 1)
        ps += __shfl_xor(ps, ofs, 32);
      rsum[r] = rsum[r] * sc + ps;
#pragma unroll
      for (int j = 0; j < 4; ++j) o[j][r] *= sc;
      Pw[m * STV + l15]      = f2bf(p0);
      Pw[m * STV + 16 + l15] = f2bf(p1);
    }
    __syncthreads();

    // ctx += P[16x32] * V[32x64]
    Frag pa;
    {
      const unsigned short* p = Pw + l15 * STV;
      pa.u[0] = *(const uint4*)(p + koff);
      pa.u[1] = *(const uint4*)(p + koff + 16);
    }
#pragma unroll
    for (int j = 0; j < 4; ++j) {
      Frag vb;
      const unsigned short* p = Vs + (j * 16 + l15) * STV;
      vb.u[0] = *(const uint4*)(p + koff);
      vb.u[1] = *(const uint4*)(p + koff + 16);
      o[j] = __builtin_amdgcn_wmma_f32_16x16x32_bf16(false, pa.v, false, vb.v, (short)0, o[j], false, false);
    }
  }

  float inv[8];
#pragma unroll
  for (int r = 0; r < 8; ++r) inv[r] = 1.0f / rsum[r];
#pragma unroll
  for (int j = 0; j < 4; ++j)
#pragma unroll
    for (int r = 0; r < 8; ++r) {
      int m = r + 8 * lhalf;
      int qrow = q0 + wave * 16 + m;
      int hd = j * 16 + l15;
      O[((size_t)bb * Sq + qrow) * Dm + hh * HDc + hd] = f2bf(o[j][r] * inv[r]);
    }
}

// =====================================================================
// Embedding: h = (x @ W + b) * sqrt(D) + PE(s, d); outputs f32 + bf16.
// =====================================================================
__global__ __launch_bounds__(256) void embed_kernel(
    const float* __restrict__ x, const float* __restrict__ W,
    const float* __restrict__ bias, float* __restrict__ outF,
    unsigned short* __restrict__ outB, int Cin)
{
  int row = blockIdx.x;
  int s = row & (Sq_ - 1);
  int tid = threadIdx.x;
  float xv[7];
#pragma unroll
  for (int c = 0; c < 7; ++c) xv[c] = (c < Cin) ? x[(size_t)row * Cin + c] : 0.0f;
#pragma unroll
  for (int t = 0; t < 2; ++t) {
    int d = tid * 2 + t;
    float acc = bias[d];
    for (int c = 0; c < Cin; ++c) acc += xv[c] * W[c * Dm + d];
    acc *= 22.627416997969522f;  // sqrt(512)
    float div = __expf((float)((d >> 1) * 2) * (-9.210340371976184f / 512.0f));
    float arg = (float)s * div;
    acc += (d & 1) ? cosf(arg) : sinf(arg);
    outF[(size_t)row * Dm + d] = acc;
    outB[(size_t)row * Dm + d] = f2bf(acc);
  }
}

// =====================================================================
// Residual + LayerNorm over D=512; outputs f32 (may alias x) + bf16.
// =====================================================================
__global__ __launch_bounds__(256) void ln_kernel(
    const float* __restrict__ x, const float* __restrict__ res,
    const float* __restrict__ w, const float* __restrict__ b,
    float* __restrict__ outF, unsigned short* __restrict__ outB)
{
  __shared__ float red[256];
  int row = blockIdx.x, tid = threadIdx.x;
  float2 xv = ((const float2*)(x   + (size_t)row * Dm))[tid];
  float2 rv = ((const float2*)(res + (size_t)row * Dm))[tid];
  float a0 = xv.x + rv.x, a1 = xv.y + rv.y;
  red[tid] = a0 + a1;
  __syncthreads();
  for (int o2 = 128; o2 > 0; o2 >>= 1) {
    if (tid < o2) red[tid] += red[tid + o2];
    __syncthreads();
  }
  float mu = red[0] * (1.0f / 512.0f);
  __syncthreads();
  float d0 = a0 - mu, d1 = a1 - mu;
  red[tid] = d0 * d0 + d1 * d1;
  __syncthreads();
  for (int o2 = 128; o2 > 0; o2 >>= 1) {
    if (tid < o2) red[tid] += red[tid + o2];
    __syncthreads();
  }
  float rstd = rsqrtf(red[0] * (1.0f / 512.0f) + 1e-5f);
  int d = tid * 2;
  float y0 = d0 * rstd * w[d] + b[d];
  float y1 = d1 * rstd * w[d + 1] + b[d + 1];
  outF[(size_t)row * Dm + d]     = y0;
  outF[(size_t)row * Dm + d + 1] = y1;
  outB[(size_t)row * Dm + d]     = f2bf(y0);
  outB[(size_t)row * Dm + d + 1] = f2bf(y1);
}

// fp32 [K,N] -> bf16 transposed [N,K]
__global__ __launch_bounds__(256) void convT_kernel(
    const float* __restrict__ W, unsigned short* __restrict__ Wt, int K, int N)
{
  int idx = blockIdx.x * 256 + threadIdx.x;
  if (idx >= K * N) return;
  int k = idx / N, n = idx - k * N;
  Wt[(size_t)n * K + k] = f2bf(W[idx]);
}

// final projection D->1: one wave per row
__global__ __launch_bounds__(256) void outproj_kernel(
    const float* __restrict__ t, const float* __restrict__ w,
    const float* __restrict__ b, float* __restrict__ out, int M)
{
  int gid = blockIdx.x * 256 + threadIdx.x;
  int row = gid >> 5;
  int lane = gid & 31;
  if (row >= M) return;
  float acc = 0.0f;
#pragma unroll
  for (int i = 0; i < 16; ++i)
    acc += t[(size_t)row * Dm + lane + i * 32] * w[lane + i * 32];
#pragma unroll
  for (int ofs = 16; ofs > 0; ofs >>= 1)
    acc += __shfl_xor(acc, ofs, 32);
  if (lane == 0) out[row] = acc + b[0];
}

// =====================================================================
// Host orchestration
// =====================================================================
extern "C" void kernel_launch(void* const* d_in, const int* in_sizes, int n_in,
                              void* d_out, int out_size, void* d_ws, size_t ws_size,
                              hipStream_t stream) {
  (void)in_sizes; (void)n_in; (void)out_size; (void)ws_size;
  const float* x_enc      = (const float*)d_in[0];
  const float* x_dec      = (const float*)d_in[1];
  const float* emb_enc_w  = (const float*)d_in[2];
  const float* emb_enc_b  = (const float*)d_in[3];
  const float* emb_dec_w  = (const float*)d_in[4];
  const float* emb_dec_b  = (const float*)d_in[5];
  const float* enc_attn_w = (const float*)d_in[6];
  const float* enc_attn_b = (const float*)d_in[7];
  const float* enc_ff_w1  = (const float*)d_in[8];
  const float* enc_ff_b1  = (const float*)d_in[9];
  const float* enc_ff_w2  = (const float*)d_in[10];
  const float* enc_ff_b2  = (const float*)d_in[11];
  const float* enc_ln_w   = (const float*)d_in[12];
  const float* enc_ln_b   = (const float*)d_in[13];
  const float* dec_self_w = (const float*)d_in[14];
  const float* dec_self_b = (const float*)d_in[15];
  const float* dec_cross_w= (const float*)d_in[16];
  const float* dec_cross_b= (const float*)d_in[17];
  const float* dec_ff_w1  = (const float*)d_in[18];
  const float* dec_ff_b1  = (const float*)d_in[19];
  const float* dec_ff_w2  = (const float*)d_in[20];
  const float* dec_ff_b2  = (const float*)d_in[21];
  const float* dec_ln_w   = (const float*)d_in[22];
  const float* dec_ln_b   = (const float*)d_in[23];
  const float* out_w      = (const float*)d_in[24];
  const float* out_b      = (const float*)d_in[25];

  const int M = Bq * Sq_;  // 8192
  char* ws = (char*)d_ws;
  size_t off = 0;
  auto alloc = [&](size_t bytes) -> void* {
    void* p = ws + off;
    off += (bytes + 255) & ~(size_t)255;
    return p;
  };
  float*          h_f   = (float*)alloc((size_t)M * Dm * 4);
  unsigned short* h_b   = (unsigned short*)alloc((size_t)M * Dm * 2);
  float*          t_f   = (float*)alloc((size_t)M * Dm * 4);
  unsigned short* t_b   = (unsigned short*)alloc((size_t)M * Dm * 2);
  unsigned short* q_b   = (unsigned short*)alloc((size_t)M * Dm * 2);
  unsigned short* k_b   = (unsigned short*)alloc((size_t)M * Dm * 2);
  unsigned short* v_b   = (unsigned short*)alloc((size_t)M * Dm * 2);
  unsigned short* ctx_b = (unsigned short*)alloc((size_t)M * Dm * 2);
  float*          g_f   = (float*)alloc((size_t)M * Dm * 4);
  unsigned short* ffn_b = (unsigned short*)alloc((size_t)M * DFFc * 2);
  unsigned short* wq  = (unsigned short*)alloc((size_t)Dm * Dm * 2);
  unsigned short* wk  = (unsigned short*)alloc((size_t)Dm * Dm * 2);
  unsigned short* wv  = (unsigned short*)alloc((size_t)Dm * Dm * 2);
  unsigned short* wo  = (unsigned short*)alloc((size_t)Dm * Dm * 2);
  unsigned short* wf1 = (unsigned short*)alloc((size_t)Dm * DFFc * 2);
  unsigned short* wf2 = (unsigned short*)alloc((size_t)DFFc * Dm * 2);

  auto convT = [&](const float* W, unsigned short* Wt, int K, int N) {
    int n = K * N;
    convT_kernel<<<(n + 255) / 256, 256, 0, stream>>>(W, Wt, K, N);
  };
  auto gemm = [&](const unsigned short* A, const unsigned short* Wt, const float* bias,
                  float* oF, unsigned short* oB, int Mm, int Nn, int Kk, int act) {
    gemm_bf16_kernel<<<dim3(Mm / 128, Nn / 128), 256, 0, stream>>>(A, Wt, bias, oF, oB, Mm, Nn, Kk, act);
  };
  auto attn = [&](const unsigned short* Qp, const unsigned short* Kp,
                  const unsigned short* Vp, unsigned short* Op) {
    banded_attn_kernel<<<dim3(Sq_ / 64, Bq * Hh), 128, 0, stream>>>(Qp, Kp, Vp, Op, Sq_, Sq_, Sq_ / 4);
  };
  auto lnorm = [&](const float* xx, const float* rr, const float* w, const float* bb,
                   float* oF, unsigned short* oB) {
    ln_kernel<<<M, 256, 0, stream>>>(xx, rr, w, bb, oF, oB);
  };

  // ----- encoder -----
  embed_kernel<<<M, 256, 0, stream>>>(x_enc, emb_enc_w, emb_enc_b, h_f, h_b, 7);
  for (int l = 0; l < 2; ++l) {
    const float* aw = enc_attn_w + (size_t)l * 4 * Dm * Dm;
    const float* ab = enc_attn_b + (size_t)l * 4 * Dm;
    convT(aw + 0 * (size_t)Dm * Dm, wq, Dm, Dm);
    convT(aw + 1 * (size_t)Dm * Dm, wk, Dm, Dm);
    convT(aw + 2 * (size_t)Dm * Dm, wv, Dm, Dm);
    convT(aw + 3 * (size_t)Dm * Dm, wo, Dm, Dm);
    gemm(h_b, wq, ab + 0 * Dm, nullptr, q_b, M, Dm, Dm, 0);
    gemm(h_b, wk, ab + 1 * Dm, nullptr, k_b, M, Dm, Dm, 0);
    gemm(h_b, wv, ab + 2 * Dm, nullptr, v_b, M, Dm, Dm, 0);
    attn(q_b, k_b, v_b, ctx_b);
    gemm(ctx_b, wo, ab + 3 * Dm, g_f, nullptr, M, Dm, Dm, 0);
    lnorm(h_f, g_f, enc_ln_w + (size_t)l * 2 * Dm, enc_ln_b + (size_t)l * 2 * Dm, h_f, h_b);
    convT(enc_ff_w1 + (size_t)l * Dm * DFFc, wf1, Dm, DFFc);
    gemm(h_b, wf1, enc_ff_b1 + (size_t)l * DFFc, nullptr, ffn_b, M, DFFc, Dm, 1);
    convT(enc_ff_w2 + (size_t)l * DFFc * Dm, wf2, DFFc, Dm);
    gemm(ffn_b, wf2, enc_ff_b2 + (size_t)l * Dm, g_f, nullptr, M, Dm, DFFc, 0);
    lnorm(h_f, g_f, enc_ln_w + (size_t)l * 2 * Dm + Dm, enc_ln_b + (size_t)l * 2 * Dm + Dm, h_f, h_b);
  }
  // h_f / h_b now hold encoder memory.

  // ----- decoder (DL = 1) -----
  embed_kernel<<<M, 256, 0, stream>>>(x_dec, emb_dec_w, emb_dec_b, t_f, t_b, 1);
  {
    // self-attention
    convT(dec_self_w + 0 * (size_t)Dm * Dm, wq, Dm, Dm);
    convT(dec_self_w + 1 * (size_t)Dm * Dm, wk, Dm, Dm);
    convT(dec_self_w + 2 * (size_t)Dm * Dm, wv, Dm, Dm);
    convT(dec_self_w + 3 * (size_t)Dm * Dm, wo, Dm, Dm);
    gemm(t_b, wq, dec_self_b + 0 * Dm, nullptr, q_b, M, Dm, Dm, 0);
    gemm(t_b, wk, dec_self_b + 1 * Dm, nullptr, k_b, M, Dm, Dm, 0);
    gemm(t_b, wv, dec_self_b + 2 * Dm, nullptr, v_b, M, Dm, Dm, 0);
    attn(q_b, k_b, v_b, ctx_b);
    gemm(ctx_b, wo, dec_self_b + 3 * Dm, g_f, nullptr, M, Dm, Dm, 0);
    lnorm(t_f, g_f, dec_ln_w + 0 * Dm, dec_ln_b + 0 * Dm, t_f, t_b);

    // cross-attention (K/V from encoder memory)
    convT(dec_cross_w + 0 * (size_t)Dm * Dm, wq, Dm, Dm);
    convT(dec_cross_w + 1 * (size_t)Dm * Dm, wk, Dm, Dm);
    convT(dec_cross_w + 2 * (size_t)Dm * Dm, wv, Dm, Dm);
    convT(dec_cross_w + 3 * (size_t)Dm * Dm, wo, Dm, Dm);
    gemm(t_b, wq, dec_cross_b + 0 * Dm, nullptr, q_b, M, Dm, Dm, 0);
    gemm(h_b, wk, dec_cross_b + 1 * Dm, nullptr, k_b, M, Dm, Dm, 0);
    gemm(h_b, wv, dec_cross_b + 2 * Dm, nullptr, v_b, M, Dm, Dm, 0);
    attn(q_b, k_b, v_b, ctx_b);
    gemm(ctx_b, wo, dec_cross_b + 3 * Dm, g_f, nullptr, M, Dm, Dm, 0);
    lnorm(t_f, g_f, dec_ln_w + 1 * Dm, dec_ln_b + 1 * Dm, t_f, t_b);

    // FFN
    convT(dec_ff_w1, wf1, Dm, DFFc);
    gemm(t_b, wf1, dec_ff_b1, nullptr, ffn_b, M, DFFc, Dm, 1);
    convT(dec_ff_w2, wf2, DFFc, Dm);
    gemm(ffn_b, wf2, dec_ff_b2, g_f, nullptr, M, Dm, DFFc, 0);
    lnorm(t_f, g_f, dec_ln_w + 2 * Dm, dec_ln_b + 2 * Dm, t_f, t_b);
  }

  outproj_kernel<<<(M * 32 + 255) / 256, 256, 0, stream>>>(t_f, out_w, out_b, (float*)d_out, M);
}
